// Colorizer_15676630630937
// MI455X (gfx1250) — matrix operand order repaired and checked by reading, not compiled, not run on hardware
//
#include <hip/hip_runtime.h>

typedef __bf16 v16bf __attribute__((ext_vector_type(16)));
typedef __bf16 v8bf  __attribute__((ext_vector_type(8)));
typedef float  v8f   __attribute__((ext_vector_type(8)));
typedef float  f4v   __attribute__((ext_vector_type(4)));

#define WAVES 4
#define KT    64            // k-tile per iteration
#define CDIM  64            // feature dim (contraction of GEMM1)
#define HW_   4096          // 64*64 spatial
#define NREF  3
#define KTOT  (NREF * HW_)  // 12288 keys
#define CY    16            // value channels
#define LPAD  8             // 72 bf16 = 144B rows: 16B-aligned, conflict-free reads

__device__ __forceinline__ v16bf cat8(v8bf lo, v8bf hi) {
  return __builtin_shufflevector(lo, hi, 0,1,2,3,4,5,6,7,8,9,10,11,12,13,14,15);
}

// balanced max over one 8-wide accumulator tile (depth 3)
__device__ __forceinline__ float tile_max8(v8f d) {
  float a = fmaxf(d[0], d[1]);
  float b = fmaxf(d[2], d[3]);
  float c = fmaxf(d[4], d[5]);
  float e = fmaxf(d[6], d[7]);
  return fmaxf(fmaxf(a, b), fmaxf(c, e));
}

__global__ __launch_bounds__(128) void colorizer_attn_kernel(
    const float* __restrict__ x, const float* __restrict__ y,
    float* __restrict__ out)
{
  __shared__ __bf16 s_ref[KT][CDIM + LPAD];        // refT tile: [k][c]
  __shared__ __bf16 s_y  [CY][KT + LPAD];          // value tile: [c'][k]
  __shared__ __bf16 s_p  [WAVES][16][KT + LPAD];   // exp(sim) tile per wave: [m][k]

  const int b     = blockIdx.y;
  const int m0wg  = blockIdx.x * (WAVES * 16);
  const int tid   = threadIdx.x;
  const int wave  = tid >> 5;
  const int lane  = tid & 31;
  const int lhalf = lane >> 4;       // 0 | 1 (lane group)
  const int nidx  = lane & 15;       // column / row index inside a 16-wide tile
  const int m     = m0wg + wave * 16 + nidx;

  // staging assignments
  const int g   = tid >> 4;          // 0..7  -> channel rows 8g..8g+7 of ref tile
  const int sub = tid & 15;          // 0..15 -> k rows 4*sub..4*sub+3
  const int cyr = tid >> 3;          // 0..15 -> value channel row
  const int p8  = tid & 7;           // 0..7  -> k offset 8*p8

  // ---- persistent B operand: target features tar[c, m] -> bf16, WMMA B layout ----
  const float* tar = x + ((size_t)(b * 4 + 3) * CDIM) * HW_;
  v16bf bq[2];
  #pragma unroll
  for (int h = 0; h < 2; ++h) {
    #pragma unroll
    for (int e = 0; e < 16; ++e) {
      int c = h * 32 + lhalf * 8 + (e & 7) + ((e >> 3) << 4);
      bq[h][e] = (__bf16)tar[(size_t)c * HW_ + m];
    }
  }

  v8f   acc     = {};
  float run_max = -__builtin_inff();
  float run_den = 0.0f;

  // ---- software-pipelined staging registers ----
  f4v rr[8];            // 8 channels x 4 k of ref tile (fp32)
  f4v ry0, ry1;         // 8 k of one value channel

  auto load_globals = [&](int t) {
    const int k0  = t * KT;
    const int r   = k0 >> 12;
    const int hw0 = k0 & (HW_ - 1);
    const float* srcr = x + ((size_t)(b * 4 + r) * CDIM + 8 * g) * HW_ + hw0 + 4 * sub;
    #pragma unroll
    for (int j = 0; j < 8; ++j)
      rr[j] = *(const f4v*)(srcr + (size_t)j * HW_);
    const float* srcy = y + ((size_t)(b * 3 + r) * CY + cyr) * HW_ + hw0 + 8 * p8;
    ry0 = *(const f4v*)(srcy);
    ry1 = *(const f4v*)(srcy + 4);
  };

  auto store_lds = [&]() {
    // ref: transpose in registers, 4 x ds_store_b128 (8 contiguous channels per row)
    #pragma unroll
    for (int kk = 0; kk < 4; ++kk) {
      v8bf pk;
      #pragma unroll
      for (int j = 0; j < 8; ++j) pk[j] = (__bf16)rr[j][kk];
      *(v8bf*)&s_ref[4 * sub + kk][8 * g] = pk;
    }
    // value tile: contiguous convert, 1 x ds_store_b128
    v8bf py;
    py[0]=(__bf16)ry0[0]; py[1]=(__bf16)ry0[1]; py[2]=(__bf16)ry0[2]; py[3]=(__bf16)ry0[3];
    py[4]=(__bf16)ry1[0]; py[5]=(__bf16)ry1[1]; py[6]=(__bf16)ry1[2]; py[7]=(__bf16)ry1[3];
    *(v8bf*)&s_y[cyr][8 * p8] = py;
  };

  const int tiles = KTOT / KT;   // 192
  load_globals(0);

  for (int t = 0; t < tiles; ++t) {
    store_lds();
    __syncthreads();
    if (t + 1 < tiles) load_globals(t + 1);   // issued now, hidden behind compute

    // ---- GEMM1: 4 sim sub-tiles (16 k-rows each), contraction C=64 = 2 WMMAs ----
    v8f d[4];
    #pragma unroll
    for (int ks = 0; ks < 4; ++ks) {
      v8f z = {};
      const int krow = ks * 16 + nidx;
      #pragma unroll
      for (int h = 0; h < 2; ++h) {
        const int c0 = h * 32 + lhalf * 8;
        v8bf a_lo = *(const v8bf*)&s_ref[krow][c0];
        v8bf a_hi = *(const v8bf*)&s_ref[krow][c0 + 16];
        v16bf a = cat8(a_lo, a_hi);
        z = __builtin_amdgcn_wmma_f32_16x16x32_bf16(false, a, false, bq[h],
                                                    (short)0, z, false, false);
      }
      d[ks] = z;
    }

    // ---- online softmax over k, per column m: balanced max tree (depth ~5) ----
    float mx0 = fmaxf(tile_max8(d[0]), tile_max8(d[1]));
    float mx1 = fmaxf(tile_max8(d[2]), tile_max8(d[3]));
    float tmax = fmaxf(mx0, mx1);
    tmax = fmaxf(tmax, __shfl_xor(tmax, 16, 32));
    const float new_max = fmaxf(run_max, tmax);
    const float scale   = __expf(run_max - new_max);
    run_max = new_max;
    #pragma unroll
    for (int v = 0; v < 8; ++v) acc[v] *= scale;

    // exp + balanced per-subtile sums + write P tile (one b128 store per sub-tile)
    float dsum[4];
    #pragma unroll
    for (int ks = 0; ks < 4; ++ks) {
      v8bf p;
      float e[8];
      #pragma unroll
      for (int v = 0; v < 8; ++v) {
        e[v] = __expf(d[ks][v] - new_max);
        p[v] = (__bf16)e[v];
      }
      dsum[ks] = ((e[0] + e[1]) + (e[2] + e[3])) + ((e[4] + e[5]) + (e[6] + e[7]));
      *(v8bf*)&s_p[wave][nidx][ks * 16 + lhalf * 8] = p;
    }
    run_den = run_den * scale + ((dsum[0] + dsum[1]) + (dsum[2] + dsum[3]));

    // ---- GEMM2: acc += yg(16 x 64) * P(64 x 16), two K=32 WMMAs ----
    #pragma unroll
    for (int h = 0; h < 2; ++h) {
      const int ka0 = h * 32 + lhalf * 8;
      v8bf ay_lo = *(const v8bf*)&s_y[nidx][ka0];
      v8bf ay_hi = *(const v8bf*)&s_y[nidx][ka0 + 16];
      v16bf ay = cat8(ay_lo, ay_hi);
      const int kb0 = h * 32 + lhalf * 8;
      v8bf bp_lo = *(const v8bf*)&s_p[wave][nidx][kb0];
      v8bf bp_hi = *(const v8bf*)&s_p[wave][nidx][kb0 + 16];
      v16bf bp = cat8(bp_lo, bp_hi);
      acc = __builtin_amdgcn_wmma_f32_16x16x32_bf16(false, ay, false, bp,
                                                    (short)0, acc, false, false);
    }
    __syncthreads();
  }

  // ---- epilogue: merge denominator across lane halves, normalize, store ----
  float den = run_den + __shfl_xor(run_den, 16, 32);
  float inv = 1.0f / den;
  float* obase = out + (size_t)b * CY * HW_ + m;
  #pragma unroll
  for (int v = 0; v < 8; ++v) {
    int crow = lhalf * 8 + v;
    obase[(size_t)crow * HW_] = acc[v] * inv;
  }
}

extern "C" void kernel_launch(void* const* d_in, const int* in_sizes, int n_in,
                              void* d_out, int out_size, void* d_ws, size_t ws_size,
                              hipStream_t stream) {
  (void)in_sizes; (void)n_in; (void)out_size; (void)d_ws; (void)ws_size;
  const float* x = (const float*)d_in[0];   // (16, 64, 64, 64) fp32
  const float* y = (const float*)d_in[1];   // (12, 16, 64, 64) fp32
  float* out = (float*)d_out;               // (4, 16, 64, 64) fp32
  dim3 grid(HW_ / (WAVES * 16), 4);         // 64 m-tiles x 4 batches
  colorizer_attn_kernel<<<grid, WAVES * 32, 0, stream>>>(x, y, out);
}
